// MultiHeadAttention_82420422410860
// MI455X (gfx1250) — matrix operand order repaired and checked by reading, not compiled, not run on hardware
//
#include <hip/hip_runtime.h>
#include <hip/hip_bf16.h>

// ---------------------------------------------------------------------------
// MHA block for MI455X (gfx1250, wave32, WMMA bf16 16x16x32, f32 accumulate)
//   q = (x @ Wq^T) * HD^-0.5 ; k = x @ Wk^T ; v = x @ Wv^T
//   scores = q k^T + bias ; attn = softmax ; ctx = attn v ; out = ctx @ Wo^T
// B=4, S=2048, D=1024, H=16, HD=64
// ---------------------------------------------------------------------------

#define B_  4
#define S_  2048
#define D_  1024
#define H_  16
#define HD_ 64

// Async global->LDS staging of V tiles (CDNA5 GLOBAL_LOAD_ASYNC_TO_LDS_B128,
// ASYNCcnt). Flip to 0 if the assembler rejects the mnemonics.
#define TRY_ASYNC 1

typedef __attribute__((ext_vector_type(16))) __bf16 v16bf;
typedef __attribute__((ext_vector_type(8)))  __bf16 v8bf;
typedef __attribute__((ext_vector_type(8)))  float  v8f;
typedef unsigned short ushort_t;

__device__ __forceinline__ ushort_t f2bf(float f) {
    union { float f; unsigned int u; } v; v.f = f;
    unsigned int u = v.u + 0x7FFFu + ((v.u >> 16) & 1u);   // round-to-nearest-even
    return (ushort_t)(u >> 16);
}

// A-fragment (16x32 bf16): lane L holds row m = L%16; element j holds
// k = (L/16)*16 + j  -> two b128 loads.
__device__ __forceinline__ v16bf load_a_frag(const ushort_t* p) {
    v8bf lo = *(const v8bf*)p;
    v8bf hi = *(const v8bf*)(p + 8);
    v16bf r;
#pragma unroll
    for (int i = 0; i < 8; ++i) { r[i] = lo[i]; r[i + 8] = hi[i]; }
    return r;
}

// B-fragment (32x16 bf16): lane L holds col n = L%16; element j holds
// k = (L/16)*16 + j -> 16 contiguous halfs = one 32B load.
__device__ __forceinline__ v16bf load_b_frag(const ushort_t* p) {
    return *(const v16bf*)p;
}

__device__ __forceinline__ v8f wmma_bf16(v16bf a, v16bf b, v8f c) {
    return __builtin_amdgcn_wmma_f32_16x16x32_bf16(false, a, false, b,
                                                   (short)0, c, false, false);
}

// ---------------------------------------------------------------------------
// Pack W (fp32 [N][K], used as x @ W^T so B(k,n) = W[n][k]) into B-fragment
// order: pack[((kb*(D/16)+nb)*32 + lane)*16 + j] = bf16(W[n][k]),
//   n = nb*16 + lane%16, k = kb*32 + (lane/16)*16 + j
// ---------------------------------------------------------------------------
__global__ __launch_bounds__(256) void pack_weight(const float* __restrict__ W,
                                                   ushort_t* __restrict__ pack) {
    int idx  = blockIdx.x * 256 + threadIdx.x;       // 0 .. D*D-1
    int j    = idx & 15;
    int lane = (idx >> 4) & 31;
    int blk  = idx >> 9;
    int nb   = blk & (D_ / 16 - 1);
    int kb   = blk >> 6;                             // D/16 == 64
    int n    = nb * 16 + (lane & 15);
    int k    = kb * 32 + (lane >> 4) * 16 + j;
    pack[idx] = f2bf(W[n * D_ + k]);
}

__global__ __launch_bounds__(256) void cvt_bf16(const float* __restrict__ x,
                                                ushort_t* __restrict__ y, int n) {
    int i = blockIdx.x * 256 + threadIdx.x;
    if (i < n) y[i] = f2bf(x[i]);
}

// ---------------------------------------------------------------------------
// GEMM: C[M=8192][N=1024] = alpha * A[M][1024] @ Bpack
// MODE 0: f32 row-major   MODE 1: bf16 row-major   MODE 2: bf16 [b][n][s]
// block = 128 threads (4 waves); wave tile 64x32 (4x2 wmma frags); WG tile 128x64
// ---------------------------------------------------------------------------
template <int MODE>
__global__ __launch_bounds__(128) void gemm_bf16(const ushort_t* __restrict__ A,
                                                 const ushort_t* __restrict__ Bp,
                                                 void* __restrict__ Cout,
                                                 float alpha) {
    const int Kd = D_, N = D_;
    int lane = threadIdx.x & 31;
    int wave = threadIdx.x >> 5;
    int half = lane >> 4;
    int l16  = lane & 15;
    int mbase = blockIdx.y * 128 + (wave >> 1) * 64;
    int nbase = blockIdx.x * 64 + (wave & 1) * 32;

    v8f c[4][2] = {};
    for (int kk = 0; kk < Kd; kk += 32) {
        v16bf a[4], b[2];
#pragma unroll
        for (int t = 0; t < 4; ++t) {
            int row = mbase + t * 16 + l16;
            a[t] = load_a_frag(A + (size_t)row * Kd + kk + half * 16);
        }
        int kbi = kk >> 5;
#pragma unroll
        for (int t = 0; t < 2; ++t) {
            int nb = (nbase + t * 16) >> 4;
            b[t] = load_b_frag(Bp + (size_t)((kbi * (N / 16) + nb) * 32 + lane) * 16);
        }
        // warm L2/L0 for the streaming operands (global_prefetch_b8)
        __builtin_prefetch(A + (size_t)(mbase + l16) * Kd + kk + 256, 0, 0);
        __builtin_prefetch(Bp + (size_t)(((kbi + 8) * (N / 16) + (nbase >> 4)) * 32 + lane) * 16, 0, 0);
#pragma unroll
        for (int i = 0; i < 4; ++i)
#pragma unroll
            for (int j = 0; j < 2; ++j)
                c[i][j] = wmma_bf16(a[i], b[j], c[i][j]);
    }

#pragma unroll
    for (int i = 0; i < 4; ++i)
#pragma unroll
        for (int j = 0; j < 2; ++j)
#pragma unroll
            for (int r = 0; r < 8; ++r) {
                int row = mbase + i * 16 + half * 8 + r;
                int col = nbase + j * 16 + l16;
                float v = c[i][j][r] * alpha;
                if (MODE == 0) {
                    ((float*)Cout)[(size_t)row * N + col] = v;
                } else if (MODE == 1) {
                    ((ushort_t*)Cout)[(size_t)row * N + col] = f2bf(v);
                } else {                       // transposed for V: [b][n][s]
                    int bb = row >> 11;        // row / S
                    int s  = row & (S_ - 1);
                    ((ushort_t*)Cout)[(size_t)(bb * D_ + col) * S_ + s] = f2bf(v);
                }
            }
}

// ---------------------------------------------------------------------------
// Flash attention: grid (B*H, S/64), 4 waves/WG, 16 query rows per wave,
// key blocks of 64. qb,kb bf16 [B*S][D]; vt bf16 [b][n=h*64+d][s]; bias f32.
// ---------------------------------------------------------------------------
__global__ __launch_bounds__(128) void attention_kernel(
    const ushort_t* __restrict__ qb, const ushort_t* __restrict__ kb,
    const ushort_t* __restrict__ vt, const float* __restrict__ bias,
    ushort_t* __restrict__ cb) {
    int bh = blockIdx.x;
    int b = bh >> 4, h = bh & 15;
    int lane = threadIdx.x & 31;
    int wave = threadIdx.x >> 5;
    int half = lane >> 4;
    int l16  = lane & 15;
    int q0 = blockIdx.y * 64 + wave * 16;

    __shared__ __align__(32) ushort_t Pt[4][16 * 64];   // P tile per wave (2KB)
#if TRY_ASYNC
    __shared__ __align__(32) ushort_t Vt[4][64 * 64];   // V tile per wave (8KB)
#endif

    // Q fragments (16 rows x 64 dims) kept in registers
    const ushort_t* qrow = qb + (size_t)(b * S_ + q0 + l16) * D_ + h * HD_;
    v16bf qa[2];
#pragma unroll
    for (int t = 0; t < 2; ++t) qa[t] = load_a_frag(qrow + t * 32 + half * 16);

    float m[8], l[8];
    v8f o[4] = {};
#pragma unroll
    for (int r = 0; r < 8; ++r) { m[r] = -INFINITY; l[r] = 0.f; }

    for (int kb0 = 0; kb0 < S_; kb0 += 64) {
#if TRY_ASYNC
        // async-stage V tile (64 dims x 64 keys bf16, [dim][key] layout):
        // 16 x GLOBAL_LOAD_ASYNC_TO_LDS_B128, tracked by ASYNCcnt
        {
            const ushort_t* vbase = vt + (size_t)(b * D_ + h * HD_) * S_ + kb0;
#pragma unroll
            for (int i = 0; i < 16; ++i) {
                int id  = i * 32 + lane;                 // 0..511 chunks of 16B
                int dim = id >> 3;
                int kc  = (id & 7) * 8;                  // key offset (8 keys)
                unsigned lds = (unsigned)(size_t)&Vt[wave][dim * 64 + kc];
                unsigned long long ga =
                    (unsigned long long)(size_t)(vbase + (size_t)dim * S_ + kc);
                asm volatile("global_load_async_to_lds_b128 %0, %1, off"
                             :: "v"(lds), "v"(ga) : "memory");
            }
        }
#endif
        // scores = Q @ K^T over 4 key chunks of 16
        v8f s[4] = {};
#pragma unroll
        for (int c = 0; c < 4; ++c)
#pragma unroll
            for (int dp = 0; dp < 2; ++dp) {
                const ushort_t* kp = kb +
                    (size_t)(b * S_ + kb0 + c * 16 + l16) * D_ + h * HD_ + dp * 32 + half * 16;
                s[c] = wmma_bf16(qa[dp], load_b_frag(kp), s[c]);
            }
        // + bias
#pragma unroll
        for (int r = 0; r < 8; ++r) {
            int qr = q0 + half * 8 + r;
            const float* bp = bias + (size_t)(b * S_ + qr) * S_ + kb0 + l16;
#pragma unroll
            for (int c = 0; c < 4; ++c) s[c][r] += bp[c * 16];
        }
        // online softmax (row stats across the 16 lanes of each half-wave)
        float al[8];
#pragma unroll
        for (int r = 0; r < 8; ++r) {
            float t = fmaxf(fmaxf(s[0][r], s[1][r]), fmaxf(s[2][r], s[3][r]));
#pragma unroll
            for (int mk = 1; mk < 16; mk <<= 1) t = fmaxf(t, __shfl_xor(t, mk, 32));
            float Mn = fmaxf(m[r], t);
            float a  = __expf(m[r] - Mn);
            float rs = 0.f;
#pragma unroll
            for (int c = 0; c < 4; ++c) {
                float p = __expf(s[c][r] - Mn);
                s[c][r] = p;
                rs += p;
            }
#pragma unroll
            for (int mk = 1; mk < 16; mk <<= 1) rs += __shfl_xor(rs, mk, 32);
            l[r] = l[r] * a + rs;
            m[r] = Mn;
            al[r] = a;
        }
#pragma unroll
        for (int c = 0; c < 4; ++c)
#pragma unroll
            for (int r = 0; r < 8; ++r) o[c][r] *= al[r];
        // P (C-layout) -> LDS -> A-layout fragments (K = 64 keys)
#pragma unroll
        for (int r = 0; r < 8; ++r) {
            int row = half * 8 + r;
#pragma unroll
            for (int c = 0; c < 4; ++c)
                Pt[wave][row * 64 + c * 16 + l16] = f2bf(s[c][r]);
        }
        __syncthreads();
        v16bf pa[2];
#pragma unroll
        for (int t = 0; t < 2; ++t)
            pa[t] = load_a_frag(&Pt[wave][l16 * 64 + t * 32 + half * 16]);
#if TRY_ASYNC
        asm volatile("s_wait_asynccnt 0x0" ::: "memory");
#endif
        // O += P @ V
#pragma unroll
        for (int c = 0; c < 4; ++c)
#pragma unroll
            for (int t = 0; t < 2; ++t) {
#if TRY_ASYNC
                const ushort_t* vp = &Vt[wave][(c * 16 + l16) * 64 + t * 32 + half * 16];
#else
                const ushort_t* vp = vt +
                    (size_t)(b * D_ + h * HD_ + c * 16 + l16) * S_ + kb0 + t * 32 + half * 16;
#endif
                o[c] = wmma_bf16(pa[t], load_b_frag(vp), o[c]);
            }
        __syncthreads();
    }

    // normalize and write context (bf16 row-major [B*S][D])
#pragma unroll
    for (int c = 0; c < 4; ++c)
#pragma unroll
        for (int r = 0; r < 8; ++r) {
            int qr = q0 + half * 8 + r;
            float v = o[c][r] / l[r];
            cb[(size_t)(b * S_ + qr) * D_ + h * HD_ + c * 16 + l16] = f2bf(v);
        }
}

// ---------------------------------------------------------------------------
extern "C" void kernel_launch(void* const* d_in, const int* in_sizes, int n_in,
                              void* d_out, int out_size, void* d_ws, size_t ws_size,
                              hipStream_t stream) {
    (void)in_sizes; (void)n_in; (void)out_size; (void)ws_size;
    const float* query = (const float*)d_in[0];
    const float* key   = (const float*)d_in[1];
    const float* value = (const float*)d_in[2];
    const float* bias  = (const float*)d_in[3];
    const float* Wq    = (const float*)d_in[4];
    const float* Wk    = (const float*)d_in[5];
    const float* Wv    = (const float*)d_in[6];
    const float* Wo    = (const float*)d_in[7];

    char* ws = (char*)d_ws;
    const size_t MB = 1ull << 20;
    ushort_t* pWq = (ushort_t*)(ws + 0 * MB);
    ushort_t* pWk = (ushort_t*)(ws + 2 * MB);
    ushort_t* pWv = (ushort_t*)(ws + 4 * MB);
    ushort_t* pWo = (ushort_t*)(ws + 6 * MB);
    ushort_t* xb  = (ushort_t*)(ws + 8 * MB);   // bf16 activations (16 MB)
    ushort_t* qbf = (ushort_t*)(ws + 24 * MB);
    ushort_t* kbf = (ushort_t*)(ws + 40 * MB);
    ushort_t* vtb = (ushort_t*)(ws + 56 * MB);
    ushort_t* cb  = xb;                          // reuse after projections

    const int nelem = B_ * S_ * D_;              // 8,388,608
    dim3 gGemm(D_ / 64, (B_ * S_) / 128);        // (16, 64)
    dim3 gAttn(B_ * H_, S_ / 64);                // (64, 32)

    pack_weight<<<D_ * D_ / 256, 256, 0, stream>>>(Wq, pWq);
    pack_weight<<<D_ * D_ / 256, 256, 0, stream>>>(Wk, pWk);
    pack_weight<<<D_ * D_ / 256, 256, 0, stream>>>(Wv, pWv);
    pack_weight<<<D_ * D_ / 256, 256, 0, stream>>>(Wo, pWo);

    cvt_bf16<<<nelem / 256, 256, 0, stream>>>(query, xb, nelem);
    gemm_bf16<1><<<gGemm, 128, 0, stream>>>(xb, pWq, qbf, 0.125f);  // * HD^-0.5

    cvt_bf16<<<nelem / 256, 256, 0, stream>>>(key, xb, nelem);
    gemm_bf16<1><<<gGemm, 128, 0, stream>>>(xb, pWk, kbf, 1.0f);

    cvt_bf16<<<nelem / 256, 256, 0, stream>>>(value, xb, nelem);
    gemm_bf16<2><<<gGemm, 128, 0, stream>>>(xb, pWv, vtb, 1.0f);

    attention_kernel<<<gAttn, 128, 0, stream>>>(qbf, kbf, vtb, bias, cb);

    gemm_bf16<0><<<gGemm, 128, 0, stream>>>(cb, pWo, d_out, 1.0f);
}
